// EEG_STDPModel_11630771437695
// MI455X (gfx1250) — compile-verified, matrix operand order checked by reading, not run on hardware
//
#include <hip/hip_runtime.h>
#include <hip/hip_bf16.h>
#include <stdint.h>
#include <math.h>

// ---------------------------------------------------------------------------
// SNN (3-layer recurrent LIF) on gfx1250 using bf16 WMMA + async LDS staging.
//   per step, per layer:  h = [in | c] @ [W ; R]^T   (K-concatenated GEMM)
//                         v = 0.5*(v + h); s = (v>=1); v *= (1-s)
// Weights repacked once per call to WMMA-B fragment-major bf16; B tiles are
// DMA'd to LDS with global_load_async_to_lds_b128 (double-buffered, ASYNCcnt),
// spikes stored bf16 row-major and read directly as A fragments.
// ---------------------------------------------------------------------------

typedef __attribute__((ext_vector_type(16))) __bf16          v16bf;
typedef __attribute__((ext_vector_type(8)))  float           v8f;
typedef __attribute__((ext_vector_type(16))) unsigned short  v16u;
typedef __attribute__((ext_vector_type(8)))  unsigned short  v8u;

constexpr int Bsz = 256;   // batch
constexpr int Lf  = 14;    // input features
constexpr int Tst = 500;   // timesteps
constexpr int Hn  = 1000;  // hidden
constexpr int HP  = 1024;  // hidden padded to 16-multiple
constexpr int NT  = HP / 16;      // 64 n-tiles of 16
constexpr int KT1 = 1 + HP / 32;  // 33 k-tiles for layer 1 (x tile + recurrent)
constexpr int KT2 = 2 * (HP / 32);// 64 k-tiles for layers 2/3

// float -> bf16 bits, round-to-nearest-even
__device__ __forceinline__ unsigned short f2bf(float f) {
  union { float f; unsigned u; } cv; cv.f = f;
  unsigned u = cv.u;
  return (unsigned short)((u + 0x7FFFu + ((u >> 16) & 1u)) >> 16);
}

// ---------------------------------------------------------------------------
// Pack [Wpart ; Rpart] (row-major [n][k] fp32) into WMMA-B fragment-major
// bf16: dst element index = ((frag*32 + lane)*16 + i), frag = kt*NT + nt,
// 16-bit B-matrix lane/element -> (k,n):
//   n_local = lane & 15
//   k_local = lane<16 ? (i<8 ? i : i+8) : (i<8 ? 8+i : 16+i)
// ---------------------------------------------------------------------------
__global__ void pack_weights(const float* __restrict__ Wsrc, int wstride, int KA,
                             int KA32, const float* __restrict__ Rsrc,
                             unsigned short* __restrict__ dst, int KTtot) {
  int idx = blockIdx.x * blockDim.x + threadIdx.x;
  int total = KTtot * NT * 512;
  if (idx >= total) return;
  int i    = idx & 15;
  int lane = (idx >> 4) & 31;
  int frag = idx >> 9;
  int nt = frag % NT;
  int kt = frag / NT;
  int half = lane >> 4;
  int klocal = half ? ((i < 8) ? 8 + i : 16 + i) : ((i < 8) ? i : i + 8);
  int k = kt * 32 + klocal;
  int n = nt * 16 + (lane & 15);
  float val = 0.0f;
  if (k < KA32) {
    if (n < Hn && k < KA) val = Wsrc[n * wstride + k];
  } else {
    int kk = k - KA32;
    if (n < Hn && kk < Hn) val = Rsrc[n * Hn + kk];
  }
  dst[idx] = f2bf(val);
}

// ---------------------------------------------------------------------------
// Fused GEMM (bf16 WMMA, fp32 accum) + LIF for one layer, one timestep.
// Block = 128 threads (4 waves). Each wave computes a 16(M) x 64(N) strip.
// B tiles (4 KB/k-tile, shared by all 4 waves) are async-DMA'd to LDS and
// double-buffered; wave w stages fragment w. Grid = (NT/4, Bsz/64) = (16,4).
// ---------------------------------------------------------------------------
__global__ void __launch_bounds__(128)
snn_layer(const float* __restrict__ x, int t,
          const unsigned short* __restrict__ s_prev,  // feed-forward spikes (bf16 [256][1024])
          const unsigned short* __restrict__ c_in,    // recurrent spikes from step t-1
          const unsigned short* __restrict__ WR,      // packed [W;R] fragment-major bf16
          float* __restrict__ v,                      // membrane fp32 [256][1024]
          unsigned short* __restrict__ s_out,         // spikes out (bf16 bits)
          int KT, int layer1) {
  __shared__ __align__(64) unsigned short Bst[2][4 * 32 * 16];  // 2 x 4 KB

  const int lane = threadIdx.x & 31;
  const int wave = threadIdx.x >> 5;
  const int half = lane >> 4;           // 0: lanes 0-15, 1: lanes 16-31
  const int m0   = blockIdx.y * 64 + wave * 16;
  const int n0   = blockIdx.x * 64;
  const int mrow = m0 + (lane & 15);
  const int ntb  = blockIdx.x * 4;      // first of 4 n-tiles

  // LDS byte offsets of this wave's staging slot (layout mirrors global pack)
  const unsigned lds0 = (unsigned)(size_t)&Bst[0][(wave * 32 + lane) * 16];
  const unsigned lds1 = (unsigned)(size_t)&Bst[1][(wave * 32 + lane) * 16];

  // Async-DMA one B fragment (wave w -> n-tile ntb+w) of k-tile kt into buf.
  // Identical 32B/lane layouts on both sides, so the shared instruction
  // offset (applied to global AND LDS address per ISA) covers the upper half.
  auto stage = [&](int kt, int buf) {
    unsigned long long ga = (unsigned long long)(size_t)
        (WR + (((size_t)kt * NT + (ntb + wave)) * 32 + lane) * 16);
    unsigned lds = buf ? lds1 : lds0;
    asm volatile(
        "global_load_async_to_lds_b128 %0, %1, off\n\t"
        "global_load_async_to_lds_b128 %0, %1, off offset:16"
        :: "v"(lds), "v"(ga) : "memory");
  };

  v8f acc0 = {0.f,0.f,0.f,0.f,0.f,0.f,0.f,0.f};
  v8f acc1 = acc0, acc2 = acc0, acc3 = acc0;

  const int khalf = KT >> 1;

  stage(0, 0);  // prologue DMA

  for (int kt = 0; kt < KT; ++kt) {
    const int cur = kt & 1;
    if (kt + 1 < KT) {
      stage(kt + 1, cur ^ 1);                            // overlap DMA w/ compute
      asm volatile("s_wait_asynccnt 0x2" ::: "memory");  // retire older group only
    } else {
      asm volatile("s_wait_asynccnt 0x0" ::: "memory");
    }
    __syncthreads();  // buf[cur] complete & visible to all waves

    // ---- A fragment (16x32 bf16, M rows = batch) ----
    v16u a;
    if (layer1 && kt == 0) {
      const float* xb = x + (size_t)mrow * Lf * Tst + t;
#pragma unroll
      for (int i = 0; i < 16; ++i) {
        int kl = half ? ((i < 8) ? 8 + i : 16 + i) : ((i < 8) ? i : i + 8);
        float val = (kl < Lf) ? xb[kl * Tst] : 0.0f;
        a[i] = f2bf(val);
      }
    } else {
      const unsigned short* src;
      int koff;
      if (layer1)            { src = c_in;   koff = (kt - 1) * 32; }
      else if (kt < khalf)   { src = s_prev; koff = kt * 32; }
      else                   { src = c_in;   koff = (kt - khalf) * 32; }
      const unsigned short* p = src + (size_t)mrow * HP + koff + half * 8;
      v8u lo = *(const v8u*)p;          // k_local {0..7} / {8..15}
      v8u hi = *(const v8u*)(p + 16);   // k_local {16..23} / {24..31}
#pragma unroll
      for (int i = 0; i < 8; ++i) { a[i] = lo[i]; a[8 + i] = hi[i]; }
    }

    // ---- 4 B fragments from LDS (ds_load_b128 pairs, 32B/lane each) ----
    const unsigned short* bb = Bst[cur];
    v16u b0 = *(const v16u*)(bb + (0 * 32 + lane) * 16);
    v16u b1 = *(const v16u*)(bb + (1 * 32 + lane) * 16);
    v16u b2 = *(const v16u*)(bb + (2 * 32 + lane) * 16);
    v16u b3 = *(const v16u*)(bb + (3 * 32 + lane) * 16);

    v16bf ab = __builtin_bit_cast(v16bf, a);
    acc0 = __builtin_amdgcn_wmma_f32_16x16x32_bf16(false, ab, false,
             __builtin_bit_cast(v16bf, b0), (short)0, acc0, false, false);
    acc1 = __builtin_amdgcn_wmma_f32_16x16x32_bf16(false, ab, false,
             __builtin_bit_cast(v16bf, b1), (short)0, acc1, false, false);
    acc2 = __builtin_amdgcn_wmma_f32_16x16x32_bf16(false, ab, false,
             __builtin_bit_cast(v16bf, b2), (short)0, acc2, false, false);
    acc3 = __builtin_amdgcn_wmma_f32_16x16x32_bf16(false, ab, false,
             __builtin_bit_cast(v16bf, b3), (short)0, acc3, false, false);

    __syncthreads();  // all reads of buf[cur] done before next DMA overwrites it
  }

  // ---- LIF epilogue: v = 0.5(v+h); s = v>=1; hard reset ----
  v8f accs[4] = {acc0, acc1, acc2, acc3};
#pragma unroll
  for (int j = 0; j < 4; ++j) {
    int n = n0 + j * 16 + (lane & 15);
#pragma unroll
    for (int r = 0; r < 8; ++r) {
      int m = m0 + (half ? 8 + r : r);
      size_t off = (size_t)m * HP + n;
      float vn = 0.5f * (v[off] + accs[j][r]);
      bool sp = (vn >= 1.0f);
      v[off] = sp ? 0.0f : vn;
      s_out[off] = sp ? (unsigned short)0x3F80 : (unsigned short)0; // bf16 1.0 / 0.0
    }
  }
}

// out = exp(v3) over the valid [256 x 1000] region
__global__ void exp_out(const float* __restrict__ v3, float* __restrict__ out) {
  int idx = blockIdx.x * blockDim.x + threadIdx.x;
  if (idx >= Bsz * Hn) return;
  int b = idx / Hn, h = idx - b * Hn;
  out[idx] = expf(v3[(size_t)b * HP + h]);
}

extern "C" void kernel_launch(void* const* d_in, const int* in_sizes, int n_in,
                              void* d_out, int out_size, void* d_ws, size_t ws_size,
                              hipStream_t stream) {
  const float* x  = (const float*)d_in[0];
  const float* W1 = (const float*)d_in[1];
  const float* R1 = (const float*)d_in[2];
  const float* W2 = (const float*)d_in[3];
  const float* R2 = (const float*)d_in[4];
  const float* W3 = (const float*)d_in[5];
  const float* R3 = (const float*)d_in[6];

  // ---- workspace layout ----
  uint8_t* ws = (uint8_t*)d_ws;
  size_t off = 0;
  unsigned short* WR1 = (unsigned short*)(ws + off); off += (size_t)KT1 * NT * 512 * 2; // 2.06 MB
  unsigned short* WR2 = (unsigned short*)(ws + off); off += (size_t)KT2 * NT * 512 * 2; // 4.0 MB
  unsigned short* WR3 = (unsigned short*)(ws + off); off += (size_t)KT2 * NT * 512 * 2; // 4.0 MB
  uint8_t* zero_base = ws + off;
  float* v1 = (float*)(ws + off); off += (size_t)Bsz * HP * 4;
  float* v2 = (float*)(ws + off); off += (size_t)Bsz * HP * 4;
  float* v3 = (float*)(ws + off); off += (size_t)Bsz * HP * 4;
  unsigned short* sb[3][2];
  for (int l = 0; l < 3; ++l)
    for (int p = 0; p < 2; ++p) {
      sb[l][p] = (unsigned short*)(ws + off);
      off += (size_t)Bsz * HP * 2;
    }
  size_t zero_bytes = (size_t)(ws + off - zero_base);

  // zero membranes + spike ping-pong buffers (t=0 reads zeros)
  hipMemsetAsync(zero_base, 0, zero_bytes, stream);

  // ---- repack weights into WMMA-B fragment-major bf16 ----
  {
    int tot1 = KT1 * NT * 512;
    int tot2 = KT2 * NT * 512;
    pack_weights<<<(tot1 + 255) / 256, 256, 0, stream>>>(W1, Lf, Lf,   32,  R1, WR1, KT1);
    pack_weights<<<(tot2 + 255) / 256, 256, 0, stream>>>(W2, Hn, Hn, HP,  R2, WR2, KT2);
    pack_weights<<<(tot2 + 255) / 256, 256, 0, stream>>>(W3, Hn, Hn, HP,  R3, WR3, KT2);
  }

  // ---- sequential LIF recurrence: 3 fused GEMM+LIF launches per step ----
  dim3 grid(NT / 4, Bsz / 64);  // (16, 4)
  dim3 blk(128);
  for (int t = 0; t < Tst; ++t) {
    int rd = t & 1, wr = rd ^ 1;
    snn_layer<<<grid, blk, 0, stream>>>(x, t, nullptr,   sb[0][rd], WR1, v1, sb[0][wr], KT1, 1);
    snn_layer<<<grid, blk, 0, stream>>>(nullptr, t, sb[0][wr], sb[1][rd], WR2, v2, sb[1][wr], KT2, 0);
    snn_layer<<<grid, blk, 0, stream>>>(nullptr, t, sb[1][wr], sb[2][rd], WR3, v3, sb[2][wr], KT2, 0);
  }

  exp_out<<<(Bsz * Hn + 255) / 256, 256, 0, stream>>>(v3, (float*)d_out);
}
